// condSequentialMonteCarlo_4690104287502
// MI455X (gfx1250) — matrix operand order repaired and checked by reading, not compiled, not run on hardware
//
#include <hip/hip_runtime.h>
#include <math.h>

// ---------------------------------------------------------------------------
// Conditional SMC forward pass for MI455X (gfx1250, wave32).
//  - GEMM1 (Z @ Omega^T, K=12)  : v_wmma_f32_16x16x4_f32  x3 (f32 precision pre-cos)
//  - GEMM2 (basis @ coeff^T, K=128): v_wmma_f32_16x16x32_f16 x4 (f32 accumulate),
//    A-operand staged through LDS and fetched as ds_load_b128 pairs.
//  - softmax/cumsum/searchsorted over 8192 particles in ONE 1024-thread
//    workgroup with the whole cumsum resident in LDS (320 KB/WGP).
//  - 200 sequential steps -> latency bound; ~600 launches, all on `stream`.
// ---------------------------------------------------------------------------

#define N_SAMPLES 8192
#define N_STEPS   200
#define DX 8
#define DU 4
#define DY 8
#define DB 128
#define SIGMA_Y 0.1f

// -0.5*DY*ln(2*pi*sigma^2) with sigma=0.1
#define LL_CONST  11.069172f
// -0.5*DX*ln(2*pi)
#define MVN_CONST (-7.3515083f)

typedef __attribute__((ext_vector_type(2)))  float    v2f;
typedef __attribute__((ext_vector_type(8)))  float    v8f;
typedef __attribute__((ext_vector_type(8)))  _Float16 v8h;
typedef __attribute__((ext_vector_type(16))) _Float16 v16h;

// ------------------------------ RNG ----------------------------------------
__device__ __forceinline__ unsigned mix32(unsigned x) {
  x ^= x >> 17; x *= 0xed5ad4bbu;
  x ^= x >> 11; x *= 0xac4c1b51u;
  x ^= x >> 15; x *= 0x31848babu;
  x ^= x >> 14; return x;
}
__device__ __forceinline__ unsigned rng_u32(unsigned seed, unsigned a, unsigned b, unsigned c) {
  unsigned h = seed * 0x9E3779B9u;
  h = mix32(h ^ (a + 0x85ebca6bu));
  h = mix32(h ^ (b * 0xc2b2ae35u + 0x27d4eb2fu));
  h = mix32(h ^ c);
  return h;
}
__device__ __forceinline__ float rng_uniform(unsigned seed, unsigned a, unsigned b, unsigned c) {
  // (0,1) open interval
  return ((float)(rng_u32(seed, a, b, c) >> 8) + 0.5f) * (1.0f / 16777216.0f);
}
__device__ __forceinline__ void rng_normal2(unsigned seed, unsigned a, unsigned b, unsigned c,
                                            float* z0, float* z1) {
  float u1 = rng_uniform(seed, a, b, 2u * c);
  float u2 = rng_uniform(seed, a, b, 2u * c + 1u);
  float r = sqrtf(-2.0f * __logf(u1));
  float s, co;
  __sincosf(6.2831853f * u2, &s, &co);
  *z0 = r * co;
  *z1 = r * s;
}

// ------------------------ small linear algebra -----------------------------
__device__ void chol8(const float* A, float* L) {
  for (int j = 0; j < 8; ++j) {
    float d = A[j * 8 + j];
    for (int k = 0; k < j; ++k) d -= L[j * 8 + k] * L[j * 8 + k];
    d = sqrtf(d);
    L[j * 8 + j] = d;
    for (int i = j + 1; i < 8; ++i) {
      float s = A[i * 8 + j];
      for (int k = 0; k < j; ++k) s -= L[i * 8 + k] * L[j * 8 + k];
      L[i * 8 + j] = s / d;
    }
    for (int k = j + 1; k < 8; ++k) L[j * 8 + k] = 0.0f;
  }
}

// K0: cholesky of init/error covariances -> consts[0..63]=L0, [64..127]=Lerr,
//     [128]=sum(log(diag(Lerr)))
__global__ void csmc_init_consts(const float* __restrict__ initCov,
                                 const float* __restrict__ errCov,
                                 float* __restrict__ consts) {
  if (threadIdx.x == 0 && blockIdx.x == 0) {
    chol8(initCov, consts);
    chol8(errCov, consts + 64);
    float sld = 0.0f;
    for (int j = 0; j < 8; ++j) sld += __logf(consts[64 + j * 8 + j]);
    consts[128] = sld;
  }
}

// K0b: state0 = mean + eps @ L0^T ; last particle pinned to ref_state[0]
__global__ __launch_bounds__(256)
void csmc_init_particles(const float* __restrict__ mean,
                         const float* __restrict__ consts,
                         const float* __restrict__ ref0,
                         float* __restrict__ state,
                         float* __restrict__ trace0,
                         float* __restrict__ log_w,
                         const int* __restrict__ seed_ptr) {
  const int i = blockIdx.x * blockDim.x + threadIdx.x;
  if (i >= N_SAMPLES) return;
  const unsigned seed = (unsigned)seed_ptr[0];
  float eps[DX];
#pragma unroll
  for (int d = 0; d < DX / 2; ++d)
    rng_normal2(seed, 0x1A17u, (unsigned)i, (unsigned)d, &eps[2 * d], &eps[2 * d + 1]);
  const float* L0 = consts;
  float s[DX];
#pragma unroll
  for (int j = 0; j < DX; ++j) {
    float v = mean[j];
#pragma unroll
    for (int k = 0; k < DX; ++k)
      if (k <= j) v += L0[j * DX + k] * eps[k];
    s[j] = v;
  }
  if (i == N_SAMPLES - 1) {
#pragma unroll
    for (int j = 0; j < DX; ++j) s[j] = ref0[j];
  }
#pragma unroll
  for (int j = 0; j < DX; ++j) {
    state[(size_t)i * DX + j]  = s[j];
    trace0[(size_t)i * DX + j] = s[j];
  }
  log_w[i] = 0.0f;
}

// ---------------------------------------------------------------------------
// K1: fused forward step. One wave = 16 particles (WMMA M-tile).
//   GEMM1: acc = Z(16x12) @ Omega^T(12x128) via 16x16x4 f32 WMMA (K=12 -> 3)
//   basis = cos(acc + phase)  -> per-wave LDS tile (f16, 16x128)
//   GEMM2: aux = basis(16x128) @ coeff^T(128x8pad16) via 16x16x32 f16 WMMA (x4)
//   epilogue (16 lanes): loglik, log_w_aux, MVN logpdf h_x (triangular solve)
// ---------------------------------------------------------------------------
__global__ __launch_bounds__(256)
void csmc_step_fwd(const float* __restrict__ state,
                   const float* __restrict__ Omega,   // (128,12) row-major
                   const float* __restrict__ phase,   // (128)
                   const float* __restrict__ coeff,   // (8,128) row-major
                   const float* __restrict__ Cm,      // (8,8)
                   const float* __restrict__ obs_t,   // (8)
                   const float* __restrict__ inp_t,   // (4)
                   const float* __restrict__ ref_t,   // (8)
                   const float* __restrict__ consts,  // L0, Lerr, sumlogdiag
                   const float* __restrict__ log_w,
                   float* __restrict__ aux_state,
                   float* __restrict__ ll_aux,
                   float* __restrict__ lwa,           // ll_aux + log_w
                   float* __restrict__ lwb) {         // lwa + h_x
  __shared__ _Float16 s_bas[8][16 * DB];  // per-wave 16x128 basis tile (32 KB)
  __shared__ float    s_aux[8][16 * DX];  // per-wave 16x8 aux tile (4 KB)

  const int lane = threadIdx.x & 31;
  const int wv   = threadIdx.x >> 5;   // wave in block, 0..7
  const int col  = lane & 15;
  const int hf   = lane >> 4;          // half-wave select
  const int m0   = (blockIdx.x * 8 + wv) * 16;   // particle tile base

  // Z row for this lane's A-matrix role (row M = col)
  float zrow[12];
  {
    const float* sp = state + (size_t)(m0 + col) * DX;
#pragma unroll
    for (int k = 0; k < DX; ++k) zrow[k] = sp[k];
#pragma unroll
    for (int k = 0; k < DU; ++k) zrow[DX + k] = inp_t[k];
  }

  // ---- GEMM1 + cos -> LDS (f16) ----
#pragma unroll 1
  for (int nt = 0; nt < DB / 16; ++nt) {
    const int n = nt * 16 + col;       // basis feature (B column / C column)
    v8f acc = {};
#pragma unroll
    for (int kb = 0; kb < 12; kb += 4) {
      v2f a, b;
      // f32 A 16x4: lanes 0-15 hold K=kb+0,kb+1 ; lanes 16-31 hold K=kb+2,kb+3
      a[0] = zrow[kb + 2 * hf];
      a[1] = zrow[kb + 2 * hf + 1];
      // f32 B 4x16 mirrors A with N across lanes; B[k][n] = Omega[n][k]
      b[0] = Omega[n * 12 + kb + 2 * hf];
      b[1] = Omega[n * 12 + kb + 2 * hf + 1];
      acc = __builtin_amdgcn_wmma_f32_16x16x4_f32(
          false, a, false, b, (short)0, acc, false, false);
    }
    const float ph = phase[n];
#pragma unroll
    for (int v = 0; v < 8; ++v) {
      // C/D layout: vgpr v, lanes 0-15 -> M=v ; lanes 16-31 -> M=v+8
      s_bas[wv][(v + 8 * hf) * DB + nt * 16 + col] = (_Float16)__cosf(acc[v] + ph);
    }
  }
  __syncthreads();

  // ---- GEMM2: basis @ coeff^T (N padded 8->16 with zeros) ----
  // B operands hoisted: contiguous 16-float runs of coeff -> f16 registers.
  v16h bb[4];
#pragma unroll
  for (int kt = 0; kt < 4; ++kt) {
#pragma unroll
    for (int i = 0; i < 16; ++i) {
      // f16 B 32x16 as used here (contiguous k, same as A): N = col, K = 16*half + i ; coeffT[K][N] = coeff[N][K]
      const int krow = kt * 32 + 16 * hf + i;
      bb[kt][i] = (col < DX) ? (_Float16)coeff[col * DB + krow] : (_Float16)0.0f;
    }
  }
  // A operand per K-tile = two contiguous 16-byte LDS chunks (ds_load_b128):
  //   a[i]   = row[kt*32 + 16*hf + i]       (i = 0..7,  VGPRs 0-3)
  //   a[8+i] = row[kt*32 + 16*hf + 8 + i]   (i = 0..7,  VGPRs 4-7)
  const _Float16* rowp = &s_bas[wv][col * DB];
  v8f acc2 = {};
#pragma unroll
  for (int kt = 0; kt < 4; ++kt) {
    const v8h lo = *(const v8h*)(rowp + kt * 32 + 16 * hf);
    const v8h hi = *(const v8h*)(rowp + kt * 32 + 16 * hf + 8);
    v16h a;
#pragma unroll
    for (int i = 0; i < 8; ++i) { a[i] = lo[i]; a[8 + i] = hi[i]; }
    acc2 = __builtin_amdgcn_wmma_f32_16x16x32_f16(
        false, a, false, bb[kt], (short)0, acc2, false, false);
  }
  if (col < DX) {
#pragma unroll
    for (int v = 0; v < 8; ++v) s_aux[wv][(v + 8 * hf) * DX + col] = acc2[v];
  }
  __syncthreads();

  // ---- per-particle epilogue: 16 lanes, one particle each ----
  if (lane < 16) {
    const int p = m0 + lane;
    float ax[DX];
#pragma unroll
    for (int j = 0; j < DX; ++j) ax[j] = s_aux[wv][lane * DX + j];
#pragma unroll
    for (int j = 0; j < DX; ++j) aux_state[(size_t)p * DX + j] = ax[j];

    // log-likelihood of obs under aux mean
    float ss = 0.0f;
#pragma unroll
    for (int y = 0; y < DY; ++y) {
      float r = obs_t[y];
#pragma unroll
      for (int k = 0; k < DX; ++k) r -= ax[k] * Cm[y * DX + k];
      ss += r * r;
    }
    const float ll = -0.5f * ss * (1.0f / (SIGMA_Y * SIGMA_Y)) + LL_CONST;
    ll_aux[p] = ll;
    const float lw = ll + log_w[p];
    lwa[p] = lw;

    // h_x: MVN logpdf of ref_t, shared chol L_err (forward substitution)
    const float* L = consts + 64;
    float sol[DX];
    float q = 0.0f;
#pragma unroll
    for (int j = 0; j < DX; ++j) {
      float s = ref_t[j] - ax[j];
#pragma unroll
      for (int k = 0; k < DX; ++k)
        if (k < j) s -= L[j * DX + k] * sol[k];
      sol[j] = s / L[j * DX + j];
      q += sol[j] * sol[j];
    }
    const float h = -0.5f * q - consts[128] + MVN_CONST;
    lwb[p] = lw + h;
  }
}

// ----------------- single-workgroup reductions over 8192 -------------------
__device__ float block_max_8192(const float* __restrict__ g, float* sred) {
  const int tid = threadIdx.x;
  float lm = -3.4e38f;
  for (int i = tid; i < N_SAMPLES; i += 1024) lm = fmaxf(lm, g[i]);
  sred[tid] = lm;
  __syncthreads();
  for (int off = 512; off > 0; off >>= 1) {
    if (tid < off) sred[tid] = fmaxf(sred[tid], sred[tid + off]);
    __syncthreads();
  }
  float m = sred[0];
  __syncthreads();
  return m;
}

// inclusive cumsum of exp(g - m) into LDS sc[8192]; returns total
__device__ float block_scan_exp(const float* __restrict__ g, float m,
                                float* sc, float* sp) {
  const int tid = threadIdx.x;
  const int base = tid * 8;
  float loc[8];
  float run = 0.0f;
#pragma unroll
  for (int j = 0; j < 8; ++j) { loc[j] = __expf(g[base + j] - m); run += loc[j]; }
  sp[tid] = run;
  __syncthreads();
  for (int off = 1; off < 1024; off <<= 1) {
    float add = (tid >= off) ? sp[tid - off] : 0.0f;
    __syncthreads();
    sp[tid] += add;
    __syncthreads();
  }
  float acc = sp[tid] - run;  // exclusive prefix for this chunk
#pragma unroll
  for (int j = 0; j < 8; ++j) { acc += loc[j]; sc[base + j] = acc; }
  float total = sp[1023];
  __syncthreads();  // also covers sc writes before any cross-thread read
  return total;
}

__device__ int lower_bound_8192(const float* sc, float v) {
  int lo = 0, hi = N_SAMPLES - 1;
  while (lo < hi) {
    int mid = (lo + hi) >> 1;
    if (sc[mid] < v) lo = mid + 1; else hi = mid;
  }
  return lo;
}

// K3: systematic resampling + reference-ancestor draw (single block, LDS scan)
__global__ __launch_bounds__(1024)
void csmc_resample(const float* __restrict__ lwa, const float* __restrict__ lwb,
                   int* __restrict__ anc_row, const int* __restrict__ seed_ptr,
                   int t) {
  __shared__ float sc[N_SAMPLES];  // 32 KB cumsum in LDS (320 KB/WGP available)
  __shared__ float sp[1024];
  const int tid = threadIdx.x;
  const unsigned seed = (unsigned)seed_ptr[0];

  const float m1 = block_max_8192(lwa, sp);
  const float t1 = block_scan_exp(lwa, m1, sc, sp);
  const float u = rng_uniform(seed, 0xA11CEu + (unsigned)t, 17u, 3u);
  const float scale = t1 * (1.0f / (float)N_SAMPLES);
  for (int i = tid; i < N_SAMPLES; i += 1024) {
    const float pos = ((float)i + u) * scale;
    anc_row[i] = lower_bound_8192(sc, pos);
  }
  __syncthreads();

  const float m2 = block_max_8192(lwb, sp);
  const float t2 = block_scan_exp(lwb, m2, sc, sp);
  if (tid == 0) {
    const float u2 = rng_uniform(seed, 0xBEEF5u + (unsigned)t, 29u, 7u);
    anc_row[N_SAMPLES - 1] = lower_bound_8192(sc, u2 * t2);  // a_idx[-1] = ref_idx
  }
}

// K4: propagate particles, pin reference, update log weights
__global__ __launch_bounds__(256)
void csmc_propagate(const float* __restrict__ aux_state,
                    const float* __restrict__ ll_aux,
                    const int* __restrict__ anc_row,
                    const float* __restrict__ consts,
                    const float* __restrict__ obs_t,
                    const float* __restrict__ ref_t,
                    const float* __restrict__ Cm,
                    float* __restrict__ state,
                    float* __restrict__ trace_t,
                    float* __restrict__ log_w,
                    const int* __restrict__ seed_ptr, int t) {
  const int i = blockIdx.x * blockDim.x + threadIdx.x;
  if (i >= N_SAMPLES) return;
  const unsigned seed = (unsigned)seed_ptr[0];

  float eps[DX];
#pragma unroll
  for (int d = 0; d < DX / 2; ++d)
    rng_normal2(seed, 0x50AD5u + (unsigned)t, (unsigned)i, (unsigned)d,
                &eps[2 * d], &eps[2 * d + 1]);

  const float* L = consts + 64;
  float ns[DX];
#pragma unroll
  for (int j = 0; j < DX; ++j) {
    float s = aux_state[(size_t)i * DX + j];
#pragma unroll
    for (int k = 0; k < DX; ++k)
      if (k <= j) s += L[j * DX + k] * eps[k];
    ns[j] = s;
  }
  if (i == N_SAMPLES - 1) {
#pragma unroll
    for (int j = 0; j < DX; ++j) ns[j] = ref_t[j];
  }
#pragma unroll
  for (int j = 0; j < DX; ++j) {
    state[(size_t)i * DX + j]   = ns[j];
    trace_t[(size_t)i * DX + j] = ns[j];
  }

  float ss = 0.0f;
#pragma unroll
  for (int y = 0; y < DY; ++y) {
    float r = obs_t[y];
#pragma unroll
    for (int k = 0; k < DX; ++k) r -= ns[k] * Cm[y * DX + k];
    ss += r * r;
  }
  const float lln = -0.5f * ss * (1.0f / (SIGMA_Y * SIGMA_Y)) + LL_CONST;
  log_w[i] = lln - ll_aux[anc_row[i]];
}

// K5: final draw + backward ancestor walk -> d_out (200 x 8)
__global__ __launch_bounds__(1024)
void csmc_finalize(const float* __restrict__ log_w,
                   const float* __restrict__ trace,
                   const int* __restrict__ anc,
                   const int* __restrict__ seed_ptr,
                   float* __restrict__ out) {
  __shared__ float sc[N_SAMPLES];
  __shared__ float sp[1024];
  const float m = block_max_8192(log_w, sp);
  const float tot = block_scan_exp(log_w, m, sc, sp);
  if (threadIdx.x == 0) {
    const unsigned seed = (unsigned)seed_ptr[0];
    const float u = rng_uniform(seed, 0xF17A1u, 5u, 11u);
    int j = lower_bound_8192(sc, u * tot);
    for (int t = N_STEPS - 1; t >= 1; --t) {
      const float* row = trace + ((size_t)t * N_SAMPLES + j) * DX;
      for (int d = 0; d < DX; ++d) out[t * DX + d] = row[d];
      j = anc[(size_t)(t - 1) * N_SAMPLES + j];
    }
    const float* row0 = trace + (size_t)j * DX;
    for (int d = 0; d < DX; ++d) out[d] = row0[d];
  }
}

// ---------------------------------------------------------------------------
extern "C" void kernel_launch(void* const* d_in, const int* in_sizes, int n_in,
                              void* d_out, int out_size, void* d_ws, size_t ws_size,
                              hipStream_t stream) {
  (void)in_sizes; (void)n_in; (void)out_size; (void)ws_size;
  const float* obs   = (const float*)d_in[0];   // (200, 8)
  const float* inp   = (const float*)d_in[1];   // (200, 4)
  const float* mean  = (const float*)d_in[2];   // (8)
  const float* icov  = (const float*)d_in[3];   // (8, 8)
  const float* coeff = (const float*)d_in[4];   // (8, 128)
  const float* ecov  = (const float*)d_in[5];   // (8, 8)
  const float* ref   = (const float*)d_in[6];   // (200, 8)
  const float* Omg   = (const float*)d_in[7];   // (128, 12)
  const float* phs   = (const float*)d_in[8];   // (128)
  const float* Cm    = (const float*)d_in[9];   // (8, 8)
  const int*   seedp = (const int*)d_in[10];    // scalar
  float* out = (float*)d_out;

  // workspace carve-up (~60 MB total; 256 B aligned slabs)
  char* w = (char*)d_ws;
  auto carve = [&](size_t bytes) {
    char* p = w;
    w += (bytes + 255) & ~(size_t)255;
    return p;
  };
  float* consts = (float*)carve(160 * sizeof(float));                         // L0, Lerr, sld
  float* state  = (float*)carve((size_t)N_SAMPLES * DX * sizeof(float));
  float* auxs   = (float*)carve((size_t)N_SAMPLES * DX * sizeof(float));
  float* llaux  = (float*)carve((size_t)N_SAMPLES * sizeof(float));
  float* lwa    = (float*)carve((size_t)N_SAMPLES * sizeof(float));
  float* lwb    = (float*)carve((size_t)N_SAMPLES * sizeof(float));
  float* logw   = (float*)carve((size_t)N_SAMPLES * sizeof(float));
  int*   anc    = (int*)carve((size_t)(N_STEPS - 1) * N_SAMPLES * sizeof(int));
  float* trace  = (float*)carve((size_t)N_STEPS * N_SAMPLES * DX * sizeof(float));

  csmc_init_consts<<<1, 64, 0, stream>>>(icov, ecov, consts);
  csmc_init_particles<<<N_SAMPLES / 256, 256, 0, stream>>>(
      mean, consts, ref, state, trace, logw, seedp);

  for (int t = 1; t < N_STEPS; ++t) {
    const float* obs_t = obs + (size_t)t * DY;
    const float* inp_t = inp + (size_t)t * DU;
    const float* ref_t = ref + (size_t)t * DX;
    int*   anc_row = anc + (size_t)(t - 1) * N_SAMPLES;
    float* trace_t = trace + (size_t)t * N_SAMPLES * DX;

    csmc_step_fwd<<<N_SAMPLES / 128, 256, 0, stream>>>(
        state, Omg, phs, coeff, Cm, obs_t, inp_t, ref_t, consts, logw,
        auxs, llaux, lwa, lwb);
    csmc_resample<<<1, 1024, 0, stream>>>(lwa, lwb, anc_row, seedp, t);
    csmc_propagate<<<N_SAMPLES / 256, 256, 0, stream>>>(
        auxs, llaux, anc_row, consts, obs_t, ref_t, Cm,
        state, trace_t, logw, seedp, t);
  }
  csmc_finalize<<<1, 1024, 0, stream>>>(logw, trace, anc, seedp, out);
}
